// DavidBeans_32547262169466
// MI455X (gfx1250) — matrix operand order repaired and checked by reading, not compiled, not run on hardware
//
#include <hip/hip_runtime.h>

typedef unsigned short u16;
typedef unsigned int   u32;
typedef __attribute__((ext_vector_type(16))) __bf16 bf16x16;
typedef __attribute__((ext_vector_type(8)))  float  f32x8;

#define BB   32
#define SS   197
#define DD   768
#define HH   12
#define PP   196
#define KNN  32
#define HDD  64
#define DFF_ 3072
#define MTOK (BB*SS)   // 6304 token rows

__device__ __forceinline__ float bf2f(u16 h) {
    return __uint_as_float(((u32)h) << 16);
}
__device__ __forceinline__ u16 f2bf(float f) {
    u32 u = __float_as_uint(f);
    u32 r = u + 0x7FFFu + ((u >> 16) & 1u);   // round-to-nearest-even
    return (u16)(r >> 16);
}

// gfx1250 async global->LDS copy (ASYNCcnt-tracked, no VGPR round trip).
// VDST = 32-bit LDS byte address, VADDR = 64-bit global address (GV mode).
__device__ __forceinline__ void async_b128_to_lds(void* lds_ptr, const void* gptr) {
    u32 loff = (u32)(uintptr_t)lds_ptr;   // low 32 bits of generic ptr == LDS offset
    asm volatile("global_load_async_to_lds_b128 %0, %1, off"
                 :: "v"(loff), "v"(gptr)
                 : "memory");
}
__device__ __forceinline__ void wait_async_le4() {
    asm volatile("s_wait_asynccnt 0x4" ::: "memory");
}
__device__ __forceinline__ void wait_async_0() {
    asm volatile("s_wait_asynccnt 0x0" ::: "memory");
}

// ---------------------------------------------------------------------------
// Weight fp32 [K][N] -> bf16 [N][K] (transposed so GEMM B-tiles are K-contiguous)
// ---------------------------------------------------------------------------
__global__ void wt_transpose(const float* __restrict__ w, u16* __restrict__ wt,
                             int K, int N) {
    int idx = blockIdx.x * 256 + threadIdx.x;
    if (idx < K * N) {
        int k = idx / N, n = idx - k * N;
        wt[(size_t)n * K + k] = f2bf(w[idx]);
    }
}

// ---------------------------------------------------------------------------
// LayerNorm (fp32 in, bf16 out). One 256-thread block per token row of 768.
// ---------------------------------------------------------------------------
__global__ __launch_bounds__(256) void layernorm_bf16out(
    const float* __restrict__ x, const float* __restrict__ g,
    const float* __restrict__ b, u16* __restrict__ out) {
    __shared__ float s1[256], s2[256];
    int row = blockIdx.x, t = threadIdx.x;
    const float* xr = x + (size_t)row * DD;
    float a0 = xr[t], a1 = xr[t + 256], a2 = xr[t + 512];
    s1[t] = a0 + a1 + a2;
    s2[t] = a0 * a0 + a1 * a1 + a2 * a2;
    __syncthreads();
    for (int o = 128; o > 0; o >>= 1) {
        if (t < o) { s1[t] += s1[t + o]; s2[t] += s2[t + o]; }
        __syncthreads();
    }
    float mu   = s1[0] * (1.0f / DD);
    float var  = s2[0] * (1.0f / DD) - mu * mu;
    float rstd = rsqrtf(var + 1e-5f);
    u16* orow = out + (size_t)row * DD;
    orow[t]       = f2bf((a0 - mu) * rstd * g[t]       + b[t]);
    orow[t + 256] = f2bf((a1 - mu) * rstd * g[t + 256] + b[t + 256]);
    orow[t + 512] = f2bf((a2 - mu) * rstd * g[t + 512] + b[t + 512]);
}

// ---------------------------------------------------------------------------
// bf16 WMMA GEMM: C[M,N] = A[M,K] @ Bt[N,K]^T + bias (+res) (gelu?) -> bf16/f32
// 64x64 tile, 128 threads (4 waves), each wave 32x32 via 2x2 wmma 16x16x32.
// Double-buffered async global->LDS staging (ASYNCcnt), branch-free row clamp.
// Fragment layouts per CDNA5 ISA 7.12.2 (bf16 A 16x32, B 32x16, f32 C 16x16).
// ---------------------------------------------------------------------------
union FragU { uint4 q[2]; bf16x16 v; };

template<bool OUT_BF16, bool DO_GELU, bool HAS_RES>
__global__ __launch_bounds__(128) void gemm_bf16_wmma(
    const u16* __restrict__ A, const u16* __restrict__ Bt,
    const float* __restrict__ bias, const float* __restrict__ res,
    void* __restrict__ Cout, int Mrows, int Ncols, int Kdim) {
    __shared__ u16 sA[2][64 * 40];   // 64 rows x 32 bf16, padded to 40 (80B pitch)
    __shared__ u16 sB[2][64 * 40];   // 64 out-cols x 32 bf16 (B pre-transposed)

    const int tid  = threadIdx.x;
    const int lane = tid & 31;
    const int wave = tid >> 5;
    const int wm = wave >> 1, wn = wave & 1;
    const int ln = lane & 15, lh = lane >> 4;
    const int rowBase = blockIdx.x * 64, colBase = blockIdx.y * 64;

    // Per-thread staging chunks: c in {tid, tid+128}; chunk -> (row, 8-elem off)
    const int r0c = tid >> 2,        o0 = (tid & 3) * 8;
    const int r1c = (tid + 128) >> 2, o1 = ((tid + 128) & 3) * 8;
    const int ga0 = min(rowBase + r0c, Mrows - 1);   // clamp: OOB rows feed only
    const int ga1 = min(rowBase + r1c, Mrows - 1);   // discarded output rows

    auto stage = [&](int k0, int bsel) {
        async_b128_to_lds(&sA[bsel][r0c * 40 + o0], A + (size_t)ga0 * Kdim + k0 + o0);
        async_b128_to_lds(&sA[bsel][r1c * 40 + o1], A + (size_t)ga1 * Kdim + k0 + o1);
        async_b128_to_lds(&sB[bsel][r0c * 40 + o0],
                          Bt + (size_t)(colBase + r0c) * Kdim + k0 + o0);
        async_b128_to_lds(&sB[bsel][r1c * 40 + o1],
                          Bt + (size_t)(colBase + r1c) * Kdim + k0 + o1);
    };

    f32x8 acc[4];
    for (int t = 0; t < 4; ++t)
        for (int e = 0; e < 8; ++e) acc[t][e] = 0.0f;

    const int nk = Kdim >> 5;
    stage(0, 0);
    for (int ic = 0; ic < nk; ++ic) {
        const int cur = ic & 1;
        if (ic + 1 < nk) {
            stage((ic + 1) << 5, cur ^ 1);   // overlap next chunk's DMA with compute
            wait_async_le4();                // 4 outstanding = just-issued next chunk
        } else {
            wait_async_0();
        }
        __syncthreads();

        // A frag (16x32 bf16): lanes 0-15 row m K={0..7,16..23}; lanes 16-31 K={8..15,24..31}
        FragU fa[2], fb[2];
        for (int i = 0; i < 2; ++i) {
            int base = (wm * 32 + i * 16 + ln) * 40 + lh * 8;
            fa[i].q[0] = *reinterpret_cast<const uint4*>(&sA[cur][base]);
            fa[i].q[1] = *reinterpret_cast<const uint4*>(&sA[cur][base + 16]);
        }
        // B frag (32x16 bf16): lanes 0-15 col n K={0..7,16..23}; lanes 16-31 K={8..15,24..31}
        for (int j = 0; j < 2; ++j) {
            int base = (wn * 32 + j * 16 + ln) * 40 + lh * 8;
            fb[j].q[0] = *reinterpret_cast<const uint4*>(&sB[cur][base]);
            fb[j].q[1] = *reinterpret_cast<const uint4*>(&sB[cur][base + 16]);
        }
        for (int i = 0; i < 2; ++i)
            for (int j = 0; j < 2; ++j)
                acc[i * 2 + j] = __builtin_amdgcn_wmma_f32_16x16x32_bf16(
                    false, fa[i].v, false, fb[j].v, (short)0, acc[i * 2 + j],
                    false, false);
        __syncthreads();
    }

    // Epilogue: C 16x16 f32 layout: VGPR jv, lanes 0-15 M=jv, lanes 16-31 M=jv+8
    for (int i = 0; i < 2; ++i)
        for (int j = 0; j < 2; ++j)
            for (int jv = 0; jv < 8; ++jv) {
                int gr = rowBase + wm * 32 + i * 16 + lh * 8 + jv;
                int gc = colBase + wn * 32 + j * 16 + ln;
                if (gr < Mrows) {
                    float v = acc[i * 2 + j][jv] + bias[gc];
                    if (HAS_RES) v += res[(size_t)gr * Ncols + gc];
                    if (DO_GELU) v = 0.5f * v * (1.0f + erff(v * 0.70710678118f));
                    if (OUT_BF16)
                        ((u16*)Cout)[(size_t)gr * Ncols + gc] = f2bf(v);
                    else
                        ((float*)Cout)[(size_t)gr * Ncols + gc] = v;
                }
            }
}

// ---------------------------------------------------------------------------
// CLS dense attention: one 256-thread block per (b,h); 197 keys.
// qkv layout: row = b*S + s, col = which*768 + h*64 + d  (bf16)
// ---------------------------------------------------------------------------
__global__ __launch_bounds__(256) void attn_cls(
    const u16* __restrict__ qkv, u16* __restrict__ out) {
    int bh = blockIdx.x;
    int b = bh / HH, h = bh - b * HH;
    int t = threadIdx.x;
    __shared__ float q[HDD];
    __shared__ float sc[256];
    __shared__ float red[256];
    const size_t rowQ = (size_t)(b * SS) * (3 * DD);
    if (t < HDD) q[t] = bf2f(qkv[rowQ + h * HDD + t]);
    __syncthreads();
    float score = -1e30f;
    if (t < SS) {
        const u16* kr = qkv + (size_t)(b * SS + t) * (3 * DD) + DD + h * HDD;
        float d = 0.f;
        for (int i = 0; i < HDD; ++i) d += q[i] * bf2f(kr[i]);
        score = d * 0.125f;   // 64^-0.5
    }
    sc[t] = score; red[t] = score;
    __syncthreads();
    for (int o = 128; o > 0; o >>= 1) { if (t < o) red[t] = fmaxf(red[t], red[t + o]); __syncthreads(); }
    float m = red[0]; __syncthreads();
    float p = (t < SS) ? expf(sc[t] - m) : 0.f;
    red[t] = p; __syncthreads();
    for (int o = 128; o > 0; o >>= 1) { if (t < o) red[t] += red[t + o]; __syncthreads(); }
    float inv = 1.0f / red[0]; __syncthreads();
    sc[t] = p * inv;
    __syncthreads();
    if (t < HDD) {
        float acc = 0.f;
        for (int j = 0; j < SS; ++j) {
            const u16* vr = qkv + (size_t)(b * SS + j) * (3 * DD) + 2 * DD + h * HDD;
            acc += sc[j] * bf2f(vr[t]);
        }
        out[(size_t)(b * SS) * DD + h * HDD + t] = f2bf(acc);
    }
}

// ---------------------------------------------------------------------------
// Sparse routed patch attention: 64-thread block per (b,p); 32 neighbors, loop h.
// Wave32 shuffle softmax across the 32 neighbor scores.
// ---------------------------------------------------------------------------
__global__ __launch_bounds__(64) void attn_patch(
    const u16* __restrict__ qkv, const int* __restrict__ routes,
    u16* __restrict__ out) {
    int bp = blockIdx.x;
    int b = bp / PP, p = bp - b * PP;
    int t = threadIdx.x;
    __shared__ int   nbr[KNN];
    __shared__ float q[HDD];
    __shared__ float w[KNN];
    if (t < KNN) nbr[t] = routes[p * KNN + t] + 1;   // +1 skips CLS
    const int qtok = b * SS + 1 + p;
    __syncthreads();
    for (int h = 0; h < HH; ++h) {
        q[t] = bf2f(qkv[(size_t)qtok * (3 * DD) + h * HDD + t]);
        __syncthreads();
        if (t < KNN) {   // wave 0 fully active -> EXEC all-1s for shuffles
            const u16* kr = qkv + (size_t)(b * SS + nbr[t]) * (3 * DD) + DD + h * HDD;
            float d = 0.f;
            for (int i = 0; i < HDD; ++i) d += q[i] * bf2f(kr[i]);
            float s = d * 0.125f;
            float m = s;
            for (int o = 16; o > 0; o >>= 1) m = fmaxf(m, __shfl_xor(m, o));
            float e = expf(s - m);
            float sum = e;
            for (int o = 16; o > 0; o >>= 1) sum += __shfl_xor(sum, o);
            w[t] = e / sum;
        }
        __syncthreads();
        float acc = 0.f;
        for (int j = 0; j < KNN; ++j) {
            const u16* vr = qkv + (size_t)(b * SS + nbr[j]) * (3 * DD) + 2 * DD + h * HDD;
            acc += w[j] * bf2f(vr[t]);
        }
        out[(size_t)qtok * DD + h * HDD + t] = f2bf(acc);
        __syncthreads();
    }
}

// ---------------------------------------------------------------------------
extern "C" void kernel_launch(void* const* d_in, const int* in_sizes, int n_in,
                              void* d_out, int out_size, void* d_ws, size_t ws_size,
                              hipStream_t stream) {
    (void)in_sizes; (void)n_in; (void)out_size; (void)ws_size;
    const float* x      = (const float*)d_in[0];
    const int*   routes = (const int*)d_in[1];
    const float* qkv_w  = (const float*)d_in[2];
    const float* qkv_b  = (const float*)d_in[3];
    const float* proj_w = (const float*)d_in[4];
    const float* proj_b = (const float*)d_in[5];
    const float* n1_g   = (const float*)d_in[6];
    const float* n1_b   = (const float*)d_in[7];
    const float* n2_g   = (const float*)d_in[8];
    const float* n2_b   = (const float*)d_in[9];
    const float* mlp_w1 = (const float*)d_in[10];
    const float* mlp_b1 = (const float*)d_in[11];
    const float* mlp_w2 = (const float*)d_in[12];
    const float* mlp_b2 = (const float*)d_in[13];

    char* ws = (char*)d_ws;
    size_t off = 0;
    auto alloc = [&](size_t bytes) -> void* {
        void* p = ws + off;
        off = (off + bytes + 255) & ~(size_t)255;
        return p;
    };
    u16*   wqkv_t  = (u16*)  alloc((size_t)DD * 3 * DD * 2);
    u16*   wproj_t = (u16*)  alloc((size_t)DD * DD * 2);
    u16*   w1_t    = (u16*)  alloc((size_t)DD * DFF_ * 2);
    u16*   w2_t    = (u16*)  alloc((size_t)DFF_ * DD * 2);
    u16*   xnorm   = (u16*)  alloc((size_t)MTOK * DD * 2);
    u16*   qkvbuf  = (u16*)  alloc((size_t)MTOK * 3 * DD * 2);
    u16*   attnout = (u16*)  alloc((size_t)MTOK * DD * 2);
    float* x1      = (float*)alloc((size_t)MTOK * DD * 4);
    u16*   x1n     = (u16*)  alloc((size_t)MTOK * DD * 2);
    u16*   hbuf    = (u16*)  alloc((size_t)MTOK * DFF_ * 2);

    // Weight convert+transpose (fp32 [K][N] -> bf16 [N][K])
    wt_transpose<<<(DD * 3 * DD + 255) / 256, 256, 0, stream>>>(qkv_w, wqkv_t, DD, 3 * DD);
    wt_transpose<<<(DD * DD + 255) / 256, 256, 0, stream>>>(proj_w, wproj_t, DD, DD);
    wt_transpose<<<(DD * DFF_ + 255) / 256, 256, 0, stream>>>(mlp_w1, w1_t, DD, DFF_);
    wt_transpose<<<(DFF_ * DD + 255) / 256, 256, 0, stream>>>(mlp_w2, w2_t, DFF_, DD);

    // LN1
    layernorm_bf16out<<<MTOK, 256, 0, stream>>>(x, n1_g, n1_b, xnorm);

    // QKV GEMM: [6304,768] x [768,2304] -> bf16
    dim3 gq((MTOK + 63) / 64, (3 * DD) / 64);
    gemm_bf16_wmma<true, false, false><<<gq, 128, 0, stream>>>(
        xnorm, wqkv_t, qkv_b, nullptr, qkvbuf, MTOK, 3 * DD, DD);

    // Attention
    attn_cls<<<BB * HH, 256, 0, stream>>>(qkvbuf, attnout);
    attn_patch<<<BB * PP, 64, 0, stream>>>(qkvbuf, routes, attnout);

    // Proj GEMM + residual(x) -> fp32 x1
    dim3 gp((MTOK + 63) / 64, DD / 64);
    gemm_bf16_wmma<false, false, true><<<gp, 128, 0, stream>>>(
        attnout, wproj_t, proj_b, x, x1, MTOK, DD, DD);

    // LN2
    layernorm_bf16out<<<MTOK, 256, 0, stream>>>(x1, n2_g, n2_b, x1n);

    // MLP1 GEMM + GELU -> bf16 h
    dim3 g1((MTOK + 63) / 64, DFF_ / 64);
    gemm_bf16_wmma<true, true, false><<<g1, 128, 0, stream>>>(
        x1n, w1_t, mlp_b1, nullptr, hbuf, MTOK, DFF_, DD);

    // MLP2 GEMM + residual(x1) -> fp32 d_out
    gemm_bf16_wmma<false, false, true><<<gp, 128, 0, stream>>>(
        hbuf, w2_t, mlp_b2, x1, d_out, MTOK, DD, DFF_);
}
